// GraphLearner_71408126263498
// MI455X (gfx1250) — matrix operand order repaired and checked by reading, not compiled, not run on hardware
//
#include <hip/hip_runtime.h>
#include <hip/hip_bf16.h>

// ---------------------------------------------------------------------------
// Fused TransformerConv graph-learner for MI455X (gfx1250, wave32, WMMA).
// B=1024 batches, N=64 nodes, C=256 in-feats, H=8 heads, O=64 per-head dim.
//
//  prep_kernel : fp32 -> bf16 weight conversion into d_ws (L2-resident).
//  fused_kernel: 1 workgroup per batch, 8 waves, 304KB dynamic LDS.
//     phase 1: Q,K,V,skip projections; ct-outer loop keeps the 8 weight
//              B-fragments (64 VGPRs) live across all 4 row tiles -> 4x less
//              L2 weight traffic (the dominant cost term).
//     phase 2: per-head scores + edge term + softmax (register softmax,
//              __shfl_xor over 16-lane halves matching C/D VGPR layout)
//     phase 3: alpha @ V (WMMA) + head mean + ecoef*ew + skip + sigmoid
//  adj^T staging uses GLOBAL_LOAD_ASYNC_TO_LDS_B32 (ASYNCcnt) when the
//  builtin exists; falls back to plain loads otherwise.
// ---------------------------------------------------------------------------

typedef __attribute__((ext_vector_type(16))) __bf16 bf16x16;
typedef __attribute__((ext_vector_type(8)))  __bf16 bf16x8;
typedef __attribute__((ext_vector_type(8)))  float  f32x8;

#define NB   1024
#define NN   64
#define NC   256
#define NH   8
#define NO   64
#define BDIM 256

// LDS layout (bytes)
#define OFF_X   0              // x bf16 [64][256]            32768
#define OFF_Q   32768          // q bf16 [64][512]            65536
#define OFF_K   98304          // k bf16 [64][512]            65536  (reused as outacc f32[64][64])
#define OFF_V   163840         // vT bf16 [8][64][64]         65536
#define OFF_A   229376         // alpha bf16 [8][64][64]      65536
#define OFF_S   294912         // skip f32 [64][64]           16384
#define SMEM_BYTES 311296
// overlays of the x region (valid after projection phase):
#define OFF_ADJT  0            // adjT f32 [64][64]           16384
#define OFF_QE    16384        // qe f32 [8][64]               2048
#define OFF_EC    18432        // ecoef f32 [8][64]            2048
#define OFF_EW    20480        // ew f32 [512]                 2048

#if defined(__HIP_DEVICE_COMPILE__) && \
    __has_builtin(__builtin_amdgcn_global_load_async_to_lds_b32) && \
    __has_builtin(__builtin_amdgcn_s_wait_asynccnt)
#define HAVE_ASYNC_LDS 1
#else
#define HAVE_ASYNC_LDS 0
#endif

#define AS_GLOBAL __attribute__((address_space(1)))
#define AS_LDS    __attribute__((address_space(3)))

__device__ __forceinline__ __bf16 f2bf(float f) {
  union { float f; unsigned u; } v; v.f = f;
  unsigned r = v.u + 0x7FFFu + ((v.u >> 16) & 1u);   // round-to-nearest-even
  union { unsigned short s; __bf16 b; } o; o.s = (unsigned short)(r >> 16);
  return o.b;
}
__device__ __forceinline__ float bfbits2f(unsigned short s) {
  union { unsigned u; float f; } v; v.u = ((unsigned)s) << 16; return v.f;
}
__device__ __forceinline__ bf16x16 cat8(bf16x8 lo, bf16x8 hi) {
  bf16x16 r;
#pragma unroll
  for (int i = 0; i < 8; ++i) { r[i] = lo[i]; r[i + 8] = hi[i]; }
  return r;
}

// A-operand 16x32 bf16 (ISA 7.12.2): lane<16 -> M=lane, K = {0..7,16..23};
// lane>=16 -> same M, K = {8..15,24..31}. Two 16-byte chunks per lane.
__device__ __forceinline__ bf16x16 frag_a(const __bf16* p, int row0, int stride, int kbase) {
  const int lane = threadIdx.x & 31;
  const __bf16* q = p + (size_t)(row0 + (lane & 15)) * stride + kbase + ((lane >> 4) << 3);
  return cat8(*(const bf16x8*)q, *(const bf16x8*)(q + 16));
}
// B-operand 32x16 bf16: lane&15 -> column N, lanes 0-15 hold K={0..7,16..23},
// lanes 16-31 hold K={8..15,24..31} -> two 16-byte chunks per lane from row N of the
// (row-major, K-contiguous) source.
__device__ __forceinline__ bf16x16 frag_b(const __bf16* p, int nrow0, int stride, int kbase) {
  const int lane = threadIdx.x & 31;
  const __bf16* q = p + (size_t)(nrow0 + (lane & 15)) * stride + kbase + ((lane >> 4) << 3);
  return cat8(*(const bf16x8*)q, *(const bf16x8*)(q + 16));
}

#define WMMA_BF16(a, b, c) \
  __builtin_amdgcn_wmma_f32_16x16x32_bf16(false, (a), false, (b), (short)0, (c), false, false)

// --------------------------- weight prep ----------------------------------
__global__ __launch_bounds__(256) void prep_kernel(const float* __restrict__ wq,
                                                   const float* __restrict__ wk,
                                                   const float* __restrict__ wv,
                                                   const float* __restrict__ ws,
                                                   __bf16* __restrict__ out) {
  int idx = blockIdx.x * BDIM + threadIdx.x;          // 1600 blocks * 256 = 409600
  if      (idx < 131072) out[idx] = f2bf(wq[idx]);
  else if (idx < 262144) out[idx] = f2bf(wk[idx - 131072]);
  else if (idx < 393216) out[idx] = f2bf(wv[idx - 262144]);
  else if (idx < 409600) out[idx] = f2bf(ws[idx - 393216]);
}

// --------------------------- fused per-batch ------------------------------
__global__ __launch_bounds__(BDIM) void fused_kernel(
    const float* __restrict__ x, const float* __restrict__ adj,
    const __bf16* __restrict__ wq, const __bf16* __restrict__ wk,
    const __bf16* __restrict__ wv, const __bf16* __restrict__ wsk,
    const float* __restrict__ bq, const float* __restrict__ bk,
    const float* __restrict__ bv, const float* __restrict__ bs,
    const float* __restrict__ we, float* __restrict__ out) {
  extern __shared__ char smem[];
  __bf16* xb   = (__bf16*)(smem + OFF_X);
  __bf16* qb   = (__bf16*)(smem + OFF_Q);
  __bf16* kb   = (__bf16*)(smem + OFF_K);
  __bf16* vtb  = (__bf16*)(smem + OFF_V);
  __bf16* alb  = (__bf16*)(smem + OFF_A);
  float*  skip = (float*)(smem + OFF_S);
  float*  adjT = (float*)(smem + OFF_ADJT);
  float*  qe   = (float*)(smem + OFF_QE);
  float*  ecoef= (float*)(smem + OFF_EC);
  float*  ewl  = (float*)(smem + OFF_EW);
  float*  outacc = (float*)(smem + OFF_K);            // reuse K region in phase 3

  const int b    = blockIdx.x;
  const int tid  = threadIdx.x;
  const int lane = tid & 31;
  // force wave-uniform wave id into an SGPR -> scalar control flow below
  const int wid  = __builtin_amdgcn_readfirstlane(tid >> 5);
  const int nlo  = lane & 15;
  const int nhi  = lane >> 4;

  // ---- stage x_b (64x256 fp32 -> bf16 LDS) ----
  const float* xg = x + (size_t)b * (NN * NC);
  for (int i = tid; i < NN * NC; i += BDIM) xb[i] = f2bf(xg[i]);
  __syncthreads();

  // ---- phase 1: Q/K/V/skip projections; reuse B-fragments across 4 mi ----
  for (int ct = wid; ct < 100; ct += 8) {             // {Q:32, K:32, V:32, skip:4}
    const __bf16* W; const float* bias; int ncol, seg;
    if      (ct < 32) { W = wq;  bias = bq; ncol = ct * 16;        seg = 0; }
    else if (ct < 64) { W = wk;  bias = bk; ncol = (ct - 32) * 16; seg = 1; }
    else if (ct < 96) { W = wv;  bias = bv; ncol = (ct - 64) * 16; seg = 2; }
    else              { W = wsk; bias = bs; ncol = (ct - 96) * 16; seg = 3; }
    // speculative prefetch of this wave's next column tile (+8 ct = +128 cols)
    __builtin_prefetch((const void*)(W + (size_t)(ncol + nlo + 128) * NC), 0, 1);
    bf16x16 bfr[8];                                    // full K=256 of weights
#pragma unroll
    for (int kt = 0; kt < 8; ++kt) bfr[kt] = frag_b(W, ncol, NC, kt * 32);
    const float bsv = bias[ncol + nlo];                // bias depends on N only
    for (int mi = 0; mi < 4; ++mi) {
      f32x8 acc;
#pragma unroll
      for (int r = 0; r < 8; ++r) acc[r] = bsv;
#pragma unroll
      for (int kt = 0; kt < 8; ++kt)
        acc = WMMA_BF16(frag_a(xb, mi * 16, NC, kt * 32), bfr[kt], acc);
      if (seg <= 1) {
        __bf16* dst = (seg == 0) ? qb : kb;
#pragma unroll
        for (int r = 0; r < 8; ++r) {
          int m = mi * 16 + r + 8 * nhi;
          dst[m * 512 + ncol + nlo] = f2bf(acc[r]);
        }
      } else if (seg == 2) {                           // V stored transposed per head
        int h2 = (ncol + nlo) >> 6, o = (ncol + nlo) & 63;
#pragma unroll
        for (int r = 0; r < 8; ++r) {
          int m = mi * 16 + r + 8 * nhi;
          vtb[(h2 * 64 + o) * 64 + m] = f2bf(acc[r]);
        }
      } else {
#pragma unroll
        for (int r = 0; r < 8; ++r) {
          int m = mi * 16 + r + 8 * nhi;
          skip[m * 64 + ncol + nlo] = acc[r];
        }
      }
    }
  }
  __syncthreads();

  // ---- stage adj^T (async gather->LDS scatter transpose) and ew ----
  const float* adjg = adj + (size_t)b * (NN * NN);
#if HAVE_ASYNC_LDS
  for (int idx = tid; idx < NN * NN; idx += BDIM) {
    int j = idx >> 6, i = idx & 63;                    // adjT[i][j] = adj[b][j][i]
    __builtin_amdgcn_global_load_async_to_lds_b32(
        (AS_GLOBAL int*)(adjg + idx),
        (AS_LDS int*)(adjT + i * 64 + j),
        0, 0);
  }
  __builtin_amdgcn_s_wait_asynccnt(0);
#else
  for (int idx = tid; idx < NN * NN; idx += BDIM) {
    int j = idx >> 6, i = idx & 63;
    adjT[i * 64 + j] = adjg[idx];
  }
#endif
  for (int idx = tid; idx < NH * NO; idx += BDIM) ewl[idx] = we[idx];
  __syncthreads();

  // ---- qe[h][i] = sum_o q[i,h,o] * ew[h,o] ----
  for (int e = tid; e < NH * NN; e += BDIM) {
    int hh = e >> 6, i = e & 63;
    const unsigned short* qrow = (const unsigned short*)qb + i * 512 + hh * 64;
    const float* ew = ewl + hh * 64;
    float s = 0.f;
#pragma unroll 8
    for (int o = 0; o < 64; ++o) s += bfbits2f(qrow[o]) * ew[o];
    qe[e] = s;
  }
  __syncthreads();

  // ---- phase 2: one head per wave; S = QK^T, scores, softmax, ecoef ----
  const int h = wid;
  for (int mi = 0; mi < 4; ++mi) {
    f32x8 sacc[4];
#pragma unroll
    for (int nj = 0; nj < 4; ++nj)
#pragma unroll
      for (int r = 0; r < 8; ++r) sacc[nj][r] = 0.f;
#pragma unroll
    for (int kt = 0; kt < 2; ++kt) {
      bf16x16 a = frag_a(qb, mi * 16, 512, h * 64 + kt * 32);
#pragma unroll
      for (int nj = 0; nj < 4; ++nj) {
        bf16x16 bb = frag_b(kb, nj * 16, 512, h * 64 + kt * 32);
        sacc[nj] = WMMA_BF16(a, bb, sacc[nj]);
      }
    }
#pragma unroll
    for (int r = 0; r < 8; ++r) {
      const int i = mi * 16 + r + 8 * nhi;             // row handled by this half-wave
      const float qv = qe[h * 64 + i];
      float av[4], sv[4];
#pragma unroll
      for (int nj = 0; nj < 4; ++nj) {
        av[nj] = adjT[i * 64 + nj * 16 + nlo];
        sv[nj] = (sacc[nj][r] + av[nj] * qv) * 0.125f; // /sqrt(O)
      }
      float mx = fmaxf(fmaxf(sv[0], sv[1]), fmaxf(sv[2], sv[3]));
#pragma unroll
      for (int m = 1; m < 16; m <<= 1) mx = fmaxf(mx, __shfl_xor(mx, m, 32));
      float ex[4], sum = 0.f;
#pragma unroll
      for (int nj = 0; nj < 4; ++nj) { ex[nj] = __expf(sv[nj] - mx); sum += ex[nj]; }
#pragma unroll
      for (int m = 1; m < 16; m <<= 1) sum += __shfl_xor(sum, m, 32);
      const float inv = 1.f / sum;
      float ec = 0.f;
#pragma unroll
      for (int nj = 0; nj < 4; ++nj) {
        float al = ex[nj] * inv;
        ec += al * av[nj];                              // sum_j alpha * adjT
        alb[(h * 64 + i) * 64 + nj * 16 + nlo] = f2bf(al);
      }
#pragma unroll
      for (int m = 1; m < 16; m <<= 1) ec += __shfl_xor(ec, m, 32);
      if (nlo == 0) ecoef[h * 64 + i] = ec;
    }
  }
  __syncthreads();

  // ---- phase 3: out = mean_h(alpha@V + ecoef*ew) + skip ----
  for (int idx = tid; idx < NN * NO; idx += BDIM) outacc[idx] = skip[idx];
  __syncthreads();
  for (int mi = 0; mi < 4; ++mi) {
    f32x8 oacc[4];
#pragma unroll
    for (int on = 0; on < 4; ++on)
#pragma unroll
      for (int r = 0; r < 8; ++r) oacc[on][r] = 0.f;
#pragma unroll
    for (int kt = 0; kt < 2; ++kt) {
      bf16x16 a = frag_a(alb + h * 4096, mi * 16, 64, kt * 32);
#pragma unroll
      for (int on = 0; on < 4; ++on) {
        bf16x16 bb = frag_b(vtb + h * 4096, on * 16, 64, kt * 32);
        oacc[on] = WMMA_BF16(a, bb, oacc[on]);
      }
    }
#pragma unroll
    for (int on = 0; on < 4; ++on)
#pragma unroll
      for (int r = 0; r < 8; ++r) {
        int i = mi * 16 + r + 8 * nhi;
        int o = on * 16 + nlo;
        float val = (oacc[on][r] + ecoef[h * 64 + i] * ewl[h * 64 + o]) * 0.125f;
        atomicAdd(&outacc[i * 64 + o], val);            // ds_add_f32
      }
  }
  __syncthreads();

  // ---- epilogue: threshold + sigmoid, coalesced store ----
  float* og = out + (size_t)b * (NN * NO);
  for (int idx = tid; idx < NN * NO; idx += BDIM) {
    float v = outacc[idx];
    og[idx] = (v > 0.1f) ? (1.f / (1.f + __expf(-v))) : 0.f;
  }
}

// --------------------------- host entry -----------------------------------
extern "C" void kernel_launch(void* const* d_in, const int* in_sizes, int n_in,
                              void* d_out, int out_size, void* d_ws, size_t ws_size,
                              hipStream_t stream) {
  (void)in_sizes; (void)n_in; (void)out_size; (void)ws_size;
  const float* x   = (const float*)d_in[0];
  const float* adj = (const float*)d_in[1];
  const float* Wq  = (const float*)d_in[2];
  const float* bq  = (const float*)d_in[3];
  const float* Wk  = (const float*)d_in[4];
  const float* bk  = (const float*)d_in[5];
  const float* Wv  = (const float*)d_in[6];
  const float* bv  = (const float*)d_in[7];
  const float* We  = (const float*)d_in[8];
  const float* Ws  = (const float*)d_in[9];
  const float* bs  = (const float*)d_in[10];
  float* out = (float*)d_out;

  __bf16* wq = (__bf16*)d_ws;          // 512*256
  __bf16* wk = wq + 512 * 256;
  __bf16* wv = wk + 512 * 256;
  __bf16* ws = wv + 512 * 256;         // 64*256; total 819200 B of workspace

  prep_kernel<<<1600, BDIM, 0, stream>>>(Wq, Wk, Wv, Ws, wq);

  (void)hipFuncSetAttribute(reinterpret_cast<const void*>(fused_kernel),
                            hipFuncAttributeMaxDynamicSharedMemorySize, SMEM_BYTES);
  fused_kernel<<<NB, BDIM, SMEM_BYTES, stream>>>(x, adj, wq, wk, wv, ws,
                                                 bq, bk, bv, bs, We, out);
}
